// GCN_5317169512694
// MI455X (gfx1250) — compile-verified, matrix operand order checked by reading, not run on hardware
//
#include <hip/hip_runtime.h>
#include <hip/hip_bf16.h>

// ---------------------------------------------------------------------------
// GCN 5-layer pipeline for MI455X (gfx1250, wave32, WMMA, async-LDS)
// - Activations -> bf16; weights -> transposed + padded bf16 (W^T[NfPad][K])
// - GEMM: block tile 128x64, 8 waves * (32x32) -> 4x v_wmma_f32_16x16x32_bf16
//   per K-step; double-buffered LDS staged via global_load_async_to_lds_b128,
//   pipelined with s_wait_asynccnt 3 (in-order async completion)
// - Aggregation: block-per-edge gather + global_atomic_add_f32 scatter
// ---------------------------------------------------------------------------

#define GCN_N 50000

typedef __attribute__((ext_vector_type(16))) __bf16 v16bf;
typedef __attribute__((ext_vector_type(8)))  float  v8f;

union FragAB { v16bf v; unsigned u[8]; };

#if defined(__has_builtin)
#if __has_builtin(__builtin_amdgcn_global_load_async_to_lds_b128) && \
    __has_builtin(__builtin_amdgcn_s_wait_asynccnt)
#define ASYNC_LDS 1
#endif
#endif
#ifndef ASYNC_LDS
#define ASYNC_LDS 0
#endif

// exact parameter types per hipcc diagnostic: int __attribute__((vector_size(16)))*
typedef int v4i_vs __attribute__((vector_size(16)));
typedef __attribute__((address_space(1))) v4i_vs* gptr_b128;
typedef __attribute__((address_space(3))) v4i_vs* lptr_b128;

__device__ __forceinline__ void g2l_b128(void* lds, const void* g) {
#if ASYNC_LDS
    __builtin_amdgcn_global_load_async_to_lds_b128(
        (gptr_b128)(unsigned long long)g,
        (lptr_b128)(unsigned)(unsigned long long)lds,
        0, 0);
#else
    *(uint4*)lds = *(const uint4*)g;
#endif
}

// wait for all but the newest 3 async ops (this thread's tile-(i+1) stages)
__device__ __forceinline__ void g2l_wait(bool more_inflight) {
#if ASYNC_LDS
    if (more_inflight) __builtin_amdgcn_s_wait_asynccnt(3);
    else               __builtin_amdgcn_s_wait_asynccnt(0);
#endif
}

__device__ __forceinline__ unsigned pack_bf16(float a, float b) {
    unsigned ua = __float_as_uint(a);
    unsigned ub = __float_as_uint(b);
    // round-to-nearest-even fp32 -> bf16
    ua = (ua + 0x7FFFu + ((ua >> 16) & 1u)) >> 16;
    ub = (ub + 0x7FFFu + ((ub >> 16) & 1u)) >> 16;
    return (ua & 0xFFFFu) | (ub << 16);
}

// --------------------------- elementwise helpers ---------------------------

__global__ void k_fill_zero(float* __restrict__ p, long n) {
    long i = (long)blockIdx.x * blockDim.x + threadIdx.x;
    long s = (long)gridDim.x * blockDim.x;
    for (; i < n; i += s) p[i] = 0.0f;
}

// fp32 -> packed bf16 pairs, row-major (n even; all activation sizes are)
__global__ void k_cvt_bf16(const float* __restrict__ in, unsigned* __restrict__ out,
                           long npairs) {
    long i = (long)blockIdx.x * blockDim.x + threadIdx.x;
    long s = (long)gridDim.x * blockDim.x;
    for (; i < npairs; i += s) out[i] = pack_bf16(in[2 * i], in[2 * i + 1]);
}

// W[K][Nf] fp32 -> W^T[NfPad][K] bf16 (pairs packed along K; pad rows zeroed)
__global__ void k_cvt_wt_bf16(const float* __restrict__ W, unsigned* __restrict__ out,
                              int K, int Nf, int NfPad) {
    const int  Kp    = K >> 1;
    const long total = (long)NfPad * Kp;
    long i = (long)blockIdx.x * blockDim.x + threadIdx.x;
    long s = (long)gridDim.x * blockDim.x;
    for (; i < total; i += s) {
        int n  = (int)(i / Kp);
        int kp = (int)(i - (long)n * Kp);
        unsigned val = 0;
        if (n < Nf)
            val = pack_bf16(W[(long)(2 * kp) * Nf + n], W[(long)(2 * kp + 1) * Nf + n]);
        out[i] = val;
    }
}

__global__ void k_deg_accum(const int* __restrict__ dst, float* __restrict__ deg, int E) {
    int i = blockIdx.x * blockDim.x + threadIdx.x;
    int s = gridDim.x * blockDim.x;
    for (; i < E; i += s) atomicAdd(&deg[dst[i]], 1.0f);
}

__global__ void k_dinv(float* __restrict__ d, int n) {
    int i = blockIdx.x * blockDim.x + threadIdx.x;
    int s = gridDim.x * blockDim.x;
    for (; i < n; i += s) d[i] = rsqrtf(d[i] + 1.0f);   // deg + self loop
}

__global__ void k_norm(const int* __restrict__ src, const int* __restrict__ dst,
                       const float* __restrict__ dinv, float* __restrict__ norm, int E) {
    int i = blockIdx.x * blockDim.x + threadIdx.x;
    int s = gridDim.x * blockDim.x;
    for (; i < E; i += s) norm[i] = dinv[src[i]] * dinv[dst[i]];
}

// ------------------------------- WMMA GEMM ---------------------------------
// C[M,Nf] = A[M,K] * W[K,Nf]; A bf16 [M][K], W^T bf16 [NfPad][K]; C fp32.
// Block 256 thr = 8 waves; block tile 128x64; wave tile 32x32 (4 WMMA/K-step).
// Double-buffered async-LDS pipeline.

__global__ void __launch_bounds__(256)
k_gemm_bf16(const unsigned short* __restrict__ Abf,
            const unsigned short* __restrict__ WbfT,
            float* __restrict__ C, int M, int K, int Nf) {
    __shared__ unsigned short lA[2][128][40];  // [buf][m][k] pitch 80B (16B align)
    __shared__ unsigned short lB[2][64][40];   // [buf][n][k] pitch 80B (16B align)

    const int t    = threadIdx.x;
    const int lane = t & 31;
    const int w    = t >> 5;
    const int l15  = lane & 15;
    const int hi   = lane >> 4;
    const int kb   = hi * 8;

    const int mBase = blockIdx.x * 128;
    const int nBase = blockIdx.y * 64;
    const int mT    = (w & 3) * 32;   // wave M sub-tile (32 rows)
    const int nT    = (w >> 2) * 32;  // wave N sub-tile (32 cols)

    // 3 async b128 stages per thread per K-step: 2 for A, 1 for B
    auto stage = [&](int buf, int k0s) {
        #pragma unroll
        for (int j = 0; j < 2; ++j) {
            int task  = t + j * 256;          // 0..511
            int rowi  = task >> 2;            // 0..127 (m)
            int chunk = task & 3;             // 4 x 16B chunks of 8 bf16
            long grow = (long)mBase + rowi;
            if (grow > M - 1) grow = M - 1;   // clamp; masked at store
            const unsigned short* gp = Abf + grow * (long)K + k0s + chunk * 8;
            g2l_b128(&lA[buf][rowi][chunk * 8], gp);
        }
        {
            int rowi  = t >> 2;               // 0..63 (n); NfPad guarantees in-range
            int chunk = t & 3;
            const unsigned short* gp = WbfT + (long)(nBase + rowi) * K + k0s + chunk * 8;
            g2l_b128(&lB[buf][rowi][chunk * 8], gp);
            if (k0s + 32 < K) __builtin_prefetch(gp + 32, 0, 1);
        }
    };

    v8f c00 = {}, c01 = {}, c10 = {}, c11 = {};

    const int nIter = K >> 5;
    stage(0, 0);
    for (int it = 0; it < nIter; ++it) {
        const int  k0   = it << 5;
        const int  cur  = it & 1;
        const bool more = (it + 1) < nIter;
        if (more) stage(cur ^ 1, k0 + 32);

        g2l_wait(more);      // oldest 3 async ops (tile `it`) complete, in order
        __syncthreads();

        // ---- fragments: packed (k,k+1) pairs -> 2x ds_load_b128 each ----
        FragAB fa0, fa1, fb0, fb1;
        #pragma unroll
        for (int v = 0; v < 8; ++v) {
            int k = (v < 4) ? (kb + 2 * v) : (kb + 16 + 2 * (v - 4));
            fa0.u[v] = *(const unsigned*)&lA[cur][mT + l15][k];
            fa1.u[v] = *(const unsigned*)&lA[cur][mT + 16 + l15][k];
            fb0.u[v] = *(const unsigned*)&lB[cur][nT + l15][k];
            fb1.u[v] = *(const unsigned*)&lB[cur][nT + 16 + l15][k];
        }

        c00 = __builtin_amdgcn_wmma_f32_16x16x32_bf16(false, fa0.v, false, fb0.v,
                                                      (short)0, c00, false, false);
        c01 = __builtin_amdgcn_wmma_f32_16x16x32_bf16(false, fa0.v, false, fb1.v,
                                                      (short)0, c01, false, false);
        c10 = __builtin_amdgcn_wmma_f32_16x16x32_bf16(false, fa1.v, false, fb0.v,
                                                      (short)0, c10, false, false);
        c11 = __builtin_amdgcn_wmma_f32_16x16x32_bf16(false, fa1.v, false, fb1.v,
                                                      (short)0, c11, false, false);

        __syncthreads();
    }

    // ---- store: VGPR r -> row (r + hi*8), col l15 within each 16x16 tile ----
    #pragma unroll
    for (int i = 0; i < 2; ++i) {
        #pragma unroll
        for (int j = 0; j < 2; ++j) {
            const v8f& cc = (i == 0) ? (j == 0 ? c00 : c01) : (j == 0 ? c10 : c11);
            int col = nBase + nT + j * 16 + l15;
            if (col < Nf) {
                #pragma unroll
                for (int r = 0; r < 8; ++r) {
                    long row = (long)mBase + mT + i * 16 + hi * 8 + r;
                    if (row < M) C[row * (long)Nf + col] = cc[r];
                }
            }
        }
    }
}

// --------------------------- edge aggregation ------------------------------
// Block per edge: src/dst/norm are block-uniform (scalar loads); lanes sweep F.

__global__ void k_agg_edge(const float* __restrict__ h, const int* __restrict__ src,
                           const int* __restrict__ dst, const float* __restrict__ norm,
                           float* __restrict__ agg, int E, int F) {
    int e = blockIdx.x;
    if (e >= E) return;
    const int   sn = src[e];
    const int   dn = dst[e];
    const float nw = norm[e];
    const float* hs = h + (long)sn * F;
    float*       ag = agg + (long)dn * F;
    for (int f = threadIdx.x; f < F; f += blockDim.x)
        atomicAdd(&ag[f], hs[f] * nw);
}

// Small-F variant (F <= 8): thread per (edge, f) via shift/mask.
__global__ void k_agg_small(const float* __restrict__ h, const int* __restrict__ src,
                            const int* __restrict__ dst, const float* __restrict__ norm,
                            float* __restrict__ agg, int E, int F, int shift) {
    long total = (long)E << shift;
    long i = (long)blockIdx.x * blockDim.x + threadIdx.x;
    long s = (long)gridDim.x * blockDim.x;
    const int mask = (1 << shift) - 1;
    for (; i < total; i += s) {
        long e = i >> shift;
        int  f = (int)i & mask;
        if (f < F) {
            float v = h[(long)src[e] * F + f] * norm[e];
            atomicAdd(&agg[(long)dst[e] * F + f], v);
        }
    }
}

// Block per row: out = (relu?)(agg + hw * dinv^2 + bias), in place on agg.
__global__ void k_finalize_row(const float* __restrict__ hw, const float* __restrict__ dinv,
                               const float* __restrict__ bias, float* __restrict__ agg,
                               int F, int do_relu) {
    const int  row  = blockIdx.x;
    const float di  = dinv[row];
    const float d2  = di * di;
    const long base = (long)row * F;
    for (int f = threadIdx.x; f < F; f += blockDim.x) {
        float v = agg[base + f] + hw[base + f] * d2 + bias[f];
        if (do_relu) v = fmaxf(v, 0.0f);
        agg[base + f] = v;
    }
}

// ------------------------------ host driver --------------------------------

static inline int cdiv(long a, long b) { return (int)((a + b - 1) / b); }
static inline int grid1d(long total, int block, long cap = 1 << 20) {
    long g = (total + block - 1) / block;
    if (g > cap) g = cap;
    if (g < 1) g = 1;
    return (int)g;
}

extern "C" void kernel_launch(void* const* d_in, const int* in_sizes, int n_in,
                              void* d_out, int out_size, void* d_ws, size_t ws_size,
                              hipStream_t stream) {
    const int Nn = GCN_N;
    const int E  = in_sizes[1] / 2;

    const float* x   = (const float*)d_in[0];
    const int*   ei  = (const int*)d_in[1];
    const int*   src = ei;
    const int*   dst = ei + E;

    const float* Wl[5] = { (const float*)d_in[2], (const float*)d_in[4],
                           (const float*)d_in[6], (const float*)d_in[8],
                           (const float*)d_in[10] };
    const float* bl[5] = { (const float*)d_in[3], (const float*)d_in[5],
                           (const float*)d_in[7], (const float*)d_in[9],
                           (const float*)d_in[11] };
    const int Fin[5]  = { 768, 1536, 768, 384, 192 };
    const int Fout[5] = { 1536, 768, 384, 192, 5 };

    float* out     = (float*)d_out;
    float* out_h   = out;                    // [N,192] layer-4 activations
    float* out_fin = out + (long)Nn * 192;   // [N,5]   layer-5 output

    // workspace layout (floats; all region offsets are 16B-aligned)
    float* dinv = (float*)d_ws;
    float* norm = dinv + Nn;
    float* bufA = norm + E;                       // GEMM result hw (N*1536)
    float* bufB = bufA + (long)Nn * 1536;         // ping
    float* bufC = bufB + (long)Nn * 1536;         // pong
    unsigned short* Abf  = (unsigned short*)(bufC + (long)Nn * 1536); // bf16 acts
    unsigned short* WbfT = Abf + (long)Nn * 1536;                     // bf16 W^T

    // ---- degree / dinv / per-edge norm (once) ----
    k_fill_zero<<<grid1d(Nn, 256), 256, 0, stream>>>(dinv, Nn);
    k_deg_accum<<<grid1d(E, 256), 256, 0, stream>>>(dst, dinv, E);
    k_dinv<<<grid1d(Nn, 256), 256, 0, stream>>>(dinv, Nn);
    k_norm<<<grid1d(E, 256), 256, 0, stream>>>(src, dst, dinv, norm, E);

    // layer schedule: input and agg/out buffers; hw always bufA
    const float* lin[5]  = { x, bufB, bufC, bufB, out_h };
    float*       lout[5] = { bufB, bufC, bufB, out_h, out_fin };

    for (int l = 0; l < 5; ++l) {
        const int  Fi    = Fin[l], Fo = Fout[l];
        const int  FoPad = (Fo + 63) & ~63;
        const long totOut = (long)Nn * Fo;

        // bf16 conversions (halve GEMM read traffic; A fits 192MB L2)
        k_cvt_bf16<<<grid1d((long)Nn * Fi / 2, 256), 256, 0, stream>>>(
            lin[l], (unsigned*)Abf, (long)Nn * Fi / 2);
        k_cvt_wt_bf16<<<grid1d((long)FoPad * Fi / 2, 256), 256, 0, stream>>>(
            Wl[l], (unsigned*)WbfT, Fi, Fo, FoPad);

        // h = in @ W   (bf16 WMMA, async double-buffered)
        dim3 ggrid(cdiv(Nn, 128), cdiv(Fo, 64));
        k_gemm_bf16<<<ggrid, 256, 0, stream>>>(Abf, WbfT, bufA, Nn, Fi, Fo);

        // agg = scatter(norm * h[src] -> dst)
        k_fill_zero<<<grid1d(totOut, 256), 256, 0, stream>>>(lout[l], totOut);
        if (Fo >= 32) {
            k_agg_edge<<<E, 256, 0, stream>>>(bufA, src, dst, norm, lout[l], E, Fo);
        } else {
            k_agg_small<<<grid1d((long)E * 8, 256), 256, 0, stream>>>(
                bufA, src, dst, norm, lout[l], E, Fo, 3);
        }

        // out = (relu?)(agg + h*dinv^2 + b)
        k_finalize_row<<<Nn, 256, 0, stream>>>(bufA, dinv, bl[l], lout[l], Fo,
                                               (l < 4) ? 1 : 0);
    }
}
